// E68SelfGatingCell_4887672783143
// MI455X (gfx1250) — matrix operand order repaired and checked by reading, not compiled, not run on hardware
//
#include <hip/hip_runtime.h>
#include <hip/hip_bf16.h>
#include <stdint.h>

// ---------------------------------------------------------------------------
// E68 self-gating cell for MI455X (gfx1250, wave32, WMMA + TDM).
//
//  split kernel : x / Wa / Wx  -> hi/lo bf16 arrays in d_ws (bf16x3 trick)
//  gemm  kernel : alpha = sigmoid(x@Wa^T+ba), v = tanh(x@Wx^T+bv)
//                 wave tile 32x32, 24 x V_WMMA_F32_16X16X32_BF16 per K-step,
//                 B panel staged in LDS by the Tensor Data Mover (double buf,
//                 s_wait_tensorcnt), A fragments straight from global bf16.
//  scan  kernel : elementwise-independent recurrence, h in a register.
//
// alpha is written into the `output` slots of d_out and v_raw into the
// h[1..T] slots; the scan overwrites them in place. Fallback path (small ws)
// does the hi/lo split on the fly.
// ---------------------------------------------------------------------------

typedef __attribute__((ext_vector_type(16))) __bf16        v16bf;
typedef __attribute__((ext_vector_type(8)))  __bf16        v8bf;
typedef __attribute__((ext_vector_type(4)))  __bf16        v4bf;
typedef __attribute__((ext_vector_type(8)))  float         v8f;
typedef __attribute__((ext_vector_type(4)))  unsigned int  u32x4;
typedef __attribute__((ext_vector_type(8)))  int           i32x8;
typedef __attribute__((ext_vector_type(4)))  int           i32x4;

#define T_DIM 2048
#define B_DIM 16
#define D_DIM 1024
#define M_DIM (T_DIM * B_DIM)          // 32768 GEMM rows
#define BD    (B_DIM * D_DIM)          // 16384 state elements

__device__ __forceinline__ float fast_sigmoid(float x) {
    return __builtin_amdgcn_rcpf(1.0f + __expf(-x));
}
__device__ __forceinline__ float fast_tanh(float x) {
    return 2.0f * __builtin_amdgcn_rcpf(1.0f + __expf(-2.0f * x)) - 1.0f;
}

__device__ __forceinline__ v8f wmma_bf16(v8f acc, v16bf a, v16bf b) {
    return __builtin_amdgcn_wmma_f32_16x16x32_bf16(false, a, false, b,
                                                   (short)0, acc, false, false);
}
// acc += A*B at ~fp32: hi*hi + hi*lo + lo*hi
__device__ __forceinline__ v8f mma3(v8f acc, v16bf ah, v16bf al,
                                    v16bf bh, v16bf bl) {
    acc = wmma_bf16(acc, ah, bh);
    acc = wmma_bf16(acc, ah, bl);
    acc = wmma_bf16(acc, al, bh);
    return acc;
}

// A fragment: per lane K = khalf*16+{0..15} (contiguous, matches B).
// p = row_base + kk + khalf*16, bf16 elements.
__device__ __forceinline__ v16bf load_a_bf16(const __bf16* __restrict__ p) {
    v8bf c0 = *(const v8bf*)(p);
    v8bf c1 = *(const v8bf*)(p + 8);
    return __builtin_shufflevector(c0, c1, 0,1,2,3,4,5,6,7,8,9,10,11,12,13,14,15);
}
// B fragment: 16 contiguous bf16 along K.
__device__ __forceinline__ v16bf load_b_bf16(const __bf16* __restrict__ p) {
    v8bf c0 = *(const v8bf*)(p);
    v8bf c1 = *(const v8bf*)(p + 8);
    return __builtin_shufflevector(c0, c1, 0,1,2,3,4,5,6,7,8,9,10,11,12,13,14,15);
}

// ---------------------------------------------------------------------------
// TDM: load a 2D bf16 tile [32 rows x 32 k] (row stride D_DIM) into LDS.
// Descriptor packing per CDNA5 ISA 8.3-8.5. data_size=1 (2B), type=2,
// count=1, tile 32x32, tensor dims D_DIM x D_DIM.
// Toolchain uses the 6-arg builtin: (g0, g1, g2, g3, g4, cpol).
// ---------------------------------------------------------------------------
__device__ __forceinline__ void tdm_load_tile(void* lds_dst,
                                              const __bf16* __restrict__ gsrc) {
    const unsigned           lds_addr = (unsigned)(uintptr_t)lds_dst;
    const unsigned long long ga       = (unsigned long long)(uintptr_t)gsrc;

    u32x4 g0;
    g0.x = 1u;                                            // count=1, user mode
    g0.y = lds_addr;                                      // lds_addr
    g0.z = (unsigned)(ga & 0xffffffffu);                  // global_addr lo
    g0.w = (unsigned)((ga >> 32) & 0x1ffffffu)            // global_addr hi
         | (2u << 30);                                    // type=2 (image)

    i32x8 g1;
    g1[0] = (1 << 16);                                    // data_size=1 (2B)
    g1[1] = (int)((D_DIM & 0xffff) << 16);                // tensor_dim0 lo16
    g1[2] = (int)(((unsigned)D_DIM >> 16)                 // tensor_dim0 hi16
         | ((D_DIM & 0xffff) << 16));                     // tensor_dim1 lo16
    g1[3] = (int)(((unsigned)D_DIM >> 16) | (32u << 16)); // dim1 hi | tile_dim0
    g1[4] = 32;                                           // tile_dim1, tile_dim2=0
    g1[5] = D_DIM;                                        // dim0_stride lo32
    g1[6] = 0;                                            // dim0_stride hi | dim1_stride lo
    g1[7] = 0;
    i32x4 gz4 = {0, 0, 0, 0};
    i32x8 gz8 = {0, 0, 0, 0, 0, 0, 0, 0};
    __builtin_amdgcn_tensor_load_to_lds(g0, g1, gz4, gz4, gz8, 0);
}

// ---------------------------------------------------------------------------
// Streaming split: f32 -> (hi bf16, lo bf16), 4 elements/thread.
// ---------------------------------------------------------------------------
__global__ __launch_bounds__(256) void e68_split_bf16(
    const float* __restrict__ src, __bf16* __restrict__ hi,
    __bf16* __restrict__ lo, int n)
{
    const int i = (blockIdx.x * 256 + threadIdx.x) * 4;
    if (i >= n) return;
    const float4 v = *(const float4*)(src + i);
    v4bf h, l;
    {
        __bf16 b;
        b = (__bf16)v.x; h[0] = b; l[0] = (__bf16)(v.x - (float)b);
        b = (__bf16)v.y; h[1] = b; l[1] = (__bf16)(v.y - (float)b);
        b = (__bf16)v.z; h[2] = b; l[2] = (__bf16)(v.z - (float)b);
        b = (__bf16)v.w; h[3] = b; l[3] = (__bf16)(v.w - (float)b);
    }
    *(v4bf*)(hi + i) = h;
    *(v4bf*)(lo + i) = l;
}

// ---------------------------------------------------------------------------
// Fast-path GEMM. Block = 8 waves, block tile 256(M) x 32(N), wave tile
// 32x32 for both outputs. B panel (2 weights x hi/lo x 32col x 32k bf16,
// 8KB) staged in LDS by TDM, double-buffered one K-step ahead.
// Grid: (M/256) * (N/32) = 128 * 32 = 4096 blocks.
// ---------------------------------------------------------------------------
__global__ __launch_bounds__(256) void e68_proj_gemm_tdm(
    const __bf16* __restrict__ Xhi, const __bf16* __restrict__ Xlo,
    const __bf16* __restrict__ Wahi, const __bf16* __restrict__ Walo,
    const __bf16* __restrict__ Wxhi, const __bf16* __restrict__ Wxlo,
    const float* __restrict__ ba,   const float* __restrict__ bv,
    float* __restrict__ alpha_out,  float* __restrict__ vraw_out)
{
    __shared__ __align__(128) __bf16 ldsb[2][4][32][32];   // [buf][slot][col][k]

    const int lane   = threadIdx.x & 31;
    const int wave   = threadIdx.x >> 5;
    const int lane16 = lane & 15;
    const int khalf  = lane >> 4;

    const int nt    = blockIdx.x & 31;
    const int mt    = blockIdx.x >> 5;
    const int nbase = nt * 32;
    const int mbase = mt * 256 + wave * 32;

    const __bf16* x0h = Xhi + (size_t)(mbase + lane16) * D_DIM;
    const __bf16* x0l = Xlo + (size_t)(mbase + lane16) * D_DIM;
    const __bf16* x1h = x0h + (size_t)16 * D_DIM;
    const __bf16* x1l = x0l + (size_t)16 * D_DIM;

    const size_t woff = (size_t)nbase * D_DIM;

    v8f acc[2][4];
#pragma unroll
    for (int a = 0; a < 2; ++a)
#pragma unroll
        for (int f = 0; f < 4; ++f) acc[a][f] = (v8f){};

    if (wave == 0) {                       // prologue: fill buffer 0, kk = 0
        tdm_load_tile(&ldsb[0][0][0][0], Wahi + woff);
        tdm_load_tile(&ldsb[0][1][0][0], Walo + woff);
        tdm_load_tile(&ldsb[0][2][0][0], Wxhi + woff);
        tdm_load_tile(&ldsb[0][3][0][0], Wxlo + woff);
    }

    for (int idx = 0; idx < 32; ++idx) {
        const int kk = idx << 5;
        if (wave == 0) {
            if (idx < 31) {                // issue next K-step into other buffer
                const size_t off = woff + kk + 32;
                const int nb = (idx + 1) & 1;
                tdm_load_tile(&ldsb[nb][0][0][0], Wahi + off);
                tdm_load_tile(&ldsb[nb][1][0][0], Walo + off);
                tdm_load_tile(&ldsb[nb][2][0][0], Wxhi + off);
                tdm_load_tile(&ldsb[nb][3][0][0], Wxlo + off);
                __builtin_amdgcn_s_wait_tensorcnt(4);   // current buf complete
            } else {
                __builtin_amdgcn_s_wait_tensorcnt(0);
            }
        }
        __syncthreads();                    // B panel visible to all waves

        const int ao = kk + khalf * 16;
        const v16bf a0h = load_a_bf16(x0h + ao);
        const v16bf a0l = load_a_bf16(x0l + ao);
        const v16bf a1h = load_a_bf16(x1h + ao);
        const v16bf a1l = load_a_bf16(x1l + ao);
        if (idx < 31) {                     // prefetch next A chunks
            __builtin_prefetch(x0h + ao + 32, 0, 3);
            __builtin_prefetch(x1h + ao + 32, 0, 3);
        }

        const __bf16* lb = &ldsb[idx & 1][0][0][0];
#pragma unroll
        for (int f = 0; f < 4; ++f) {
            const int shi = (f >> 1) * 2;          // 0: Wa, 2: Wx
            const int col = (f & 1) * 16 + lane16; // n-tile 0 / 1
            const __bf16* ph = lb + (size_t)(shi + 0) * 1024 + col * 32 + khalf * 16;
            const __bf16* pl = lb + (size_t)(shi + 1) * 1024 + col * 32 + khalf * 16;
            const v16bf bh = load_b_bf16(ph);
            const v16bf bl = load_b_bf16(pl);
            acc[0][f] = mma3(acc[0][f], a0h, a0l, bh, bl);
            acc[1][f] = mma3(acc[1][f], a1h, a1l, bh, bl);
        }
        __syncthreads();                    // reads done before buffer reuse
    }

    const float bia0 = ba[nbase + lane16];
    const float bia1 = ba[nbase + 16 + lane16];
    const float biv0 = bv[nbase + lane16];
    const float biv1 = bv[nbase + 16 + lane16];

#pragma unroll
    for (int a = 0; a < 2; ++a) {
        const int rowb = mbase + a * 16 + khalf * 8;   // C: elem r -> M=8*khalf+r
#pragma unroll
        for (int r = 0; r < 8; ++r) {
            const size_t o = (size_t)(rowb + r) * D_DIM + nbase + lane16;
            alpha_out[o]      = fast_sigmoid(acc[a][0][r] + bia0);
            alpha_out[o + 16] = fast_sigmoid(acc[a][1][r] + bia1);
            vraw_out[o]       = fast_tanh(acc[a][2][r] + biv0);
            vraw_out[o + 16]  = fast_tanh(acc[a][3][r] + biv1);
        }
    }
}

// ---------------------------------------------------------------------------
// Fallback GEMM (small d_ws): round-1 on-the-fly split, direct loads.
// ---------------------------------------------------------------------------
struct Frag { v16bf hi; v16bf lo; };

__device__ __forceinline__ Frag load_split_b(const float* __restrict__ p) {
    Frag f;
#pragma unroll
    for (int e = 0; e < 16; ++e) {
        float v = p[e]; __bf16 h = (__bf16)v;
        f.hi[e] = h; f.lo[e] = (__bf16)(v - (float)h);
    }
    return f;
}
__device__ __forceinline__ Frag load_split_a(const float* __restrict__ p) {
    Frag f;
#pragma unroll
    for (int e = 0; e < 8; ++e) {
        float v = p[e]; __bf16 h = (__bf16)v;
        f.hi[e] = h; f.lo[e] = (__bf16)(v - (float)h);
    }
#pragma unroll
    for (int e = 0; e < 8; ++e) {
        float v = p[8 + e]; __bf16 h = (__bf16)v;
        f.hi[8+e] = h; f.lo[8+e] = (__bf16)(v - (float)h);
    }
    return f;
}

__global__ __launch_bounds__(256) void e68_proj_gemm_fallback(
    const float* __restrict__ X,  const float* __restrict__ Wa,
    const float* __restrict__ ba, const float* __restrict__ Wx,
    const float* __restrict__ bv,
    float* __restrict__ alpha_out, float* __restrict__ vraw_out)
{
    const int lane   = threadIdx.x & 31;
    const int wave   = threadIdx.x >> 5;
    const int lane16 = lane & 15;
    const int khalf  = lane >> 4;

    const int nt    = blockIdx.x & 31;
    const int mt    = blockIdx.x >> 5;
    const int mbase = mt * 128 + wave * 16;
    const int nbase = nt * 32;

    const float* xrow = X  + (size_t)(mbase + lane16)      * D_DIM;
    const float* wa0  = Wa + (size_t)(nbase + lane16)      * D_DIM;
    const float* wa1  = Wa + (size_t)(nbase + 16 + lane16) * D_DIM;
    const float* wx0  = Wx + (size_t)(nbase + lane16)      * D_DIM;
    const float* wx1  = Wx + (size_t)(nbase + 16 + lane16) * D_DIM;

    v8f acc_a0 = {}; v8f acc_a1 = {}; v8f acc_v0 = {}; v8f acc_v1 = {};

    for (int kk = 0; kk < D_DIM; kk += 32) {
        const Frag A = load_split_a(xrow + kk + khalf * 16);
        const Frag Ba0 = load_split_b(wa0 + kk + khalf * 16);
        acc_a0 = mma3(acc_a0, A.hi, A.lo, Ba0.hi, Ba0.lo);
        const Frag Ba1 = load_split_b(wa1 + kk + khalf * 16);
        acc_a1 = mma3(acc_a1, A.hi, A.lo, Ba1.hi, Ba1.lo);
        const Frag Bx0 = load_split_b(wx0 + kk + khalf * 16);
        acc_v0 = mma3(acc_v0, A.hi, A.lo, Bx0.hi, Bx0.lo);
        const Frag Bx1 = load_split_b(wx1 + kk + khalf * 16);
        acc_v1 = mma3(acc_v1, A.hi, A.lo, Bx1.hi, Bx1.lo);
    }

    const float bia0 = ba[nbase + lane16];
    const float bia1 = ba[nbase + 16 + lane16];
    const float biv0 = bv[nbase + lane16];
    const float biv1 = bv[nbase + 16 + lane16];
#pragma unroll
    for (int r = 0; r < 8; ++r) {
        const size_t o = (size_t)(mbase + khalf * 8 + r) * D_DIM + nbase + lane16;
        alpha_out[o]      = fast_sigmoid(acc_a0[r] + bia0);
        alpha_out[o + 16] = fast_sigmoid(acc_a1[r] + bia1);
        vraw_out[o]       = fast_tanh(acc_v0[r] + biv0);
        vraw_out[o + 16]  = fast_tanh(acc_v1[r] + biv1);
    }
}

// ---------------------------------------------------------------------------
// Recurrent scan: 16384 independent scans, h register-resident.
// ---------------------------------------------------------------------------
__global__ __launch_bounds__(256) void e68_scan(
    const float* __restrict__ h0, const float* __restrict__ d_g,
    const float* __restrict__ b_g,
    float* __restrict__ out_buf,   // [T, BD]   in: alpha, out: h*silu(h)
    float* __restrict__ h_buf)     // [T+1, BD] in: [1..T]=v_raw, out: h
{
    const int   i  = blockIdx.x * blockDim.x + threadIdx.x;
    const int   d  = i & (D_DIM - 1);
    const float dg = d_g[d];
    const float bg = b_g[d];

    float h = h0[i];
    h_buf[i] = h;

    size_t po = (size_t)i;
    size_t ph = (size_t)BD + i;
    for (int t = 0; t < T_DIM; ++t) {
        const float alpha = out_buf[po];
        const float vraw  = h_buf[ph];
        const float g = fast_sigmoid(dg * h + bg);
        h = alpha * h + (1.0f - alpha) * (vraw * g);
        out_buf[po] = h * h * fast_sigmoid(h);
        h_buf[ph]   = h;
        po += BD; ph += BD;
    }
}

// ---------------------------------------------------------------------------
extern "C" void kernel_launch(void* const* d_in, const int* in_sizes, int n_in,
                              void* d_out, int out_size, void* d_ws, size_t ws_size,
                              hipStream_t stream) {
    (void)in_sizes; (void)n_in; (void)out_size;

    const float* x  = (const float*)d_in[0];
    const float* h0 = (const float*)d_in[1];
    const float* Wa = (const float*)d_in[2];
    const float* ba = (const float*)d_in[3];
    const float* Wx = (const float*)d_in[4];
    const float* bv = (const float*)d_in[5];
    const float* dg = (const float*)d_in[6];
    const float* bg = (const float*)d_in[7];

    float* out_buf = (float*)d_out;
    float* h_buf   = out_buf + (size_t)T_DIM * BD;
    float* vraw    = h_buf + BD;

    const size_t MD = (size_t)M_DIM * D_DIM;     // 33.5M
    const size_t DD = (size_t)D_DIM * D_DIM;     // 1M
    const size_t need = (2 * MD + 4 * DD) * sizeof(__bf16);

    if (ws_size >= need) {
        __bf16* xhi  = (__bf16*)d_ws;
        __bf16* xlo  = xhi  + MD;
        __bf16* wahi = xlo  + MD;
        __bf16* walo = wahi + DD;
        __bf16* wxhi = walo + DD;
        __bf16* wxlo = wxhi + DD;

        e68_split_bf16<<<(int)(MD / 1024), 256, 0, stream>>>(x,  xhi,  xlo,  (int)MD);
        e68_split_bf16<<<(int)(DD / 1024), 256, 0, stream>>>(Wa, wahi, walo, (int)DD);
        e68_split_bf16<<<(int)(DD / 1024), 256, 0, stream>>>(Wx, wxhi, wxlo, (int)DD);

        const int blocks = (M_DIM / 256) * (D_DIM / 32);   // 4096
        e68_proj_gemm_tdm<<<blocks, 256, 0, stream>>>(xhi, xlo, wahi, walo,
                                                      wxhi, wxlo, ba, bv,
                                                      out_buf, vraw);
    } else {
        const int blocks = (M_DIM / 128) * (D_DIM / 32);   // 8192
        e68_proj_gemm_fallback<<<blocks, 256, 0, stream>>>(x, Wa, ba, Wx, bv,
                                                           out_buf, vraw);
    }

    e68_scan<<<BD / 256, 256, 0, stream>>>(h0, dg, bg, out_buf, h_buf);
}
